// VectorSampling_137438954202
// MI455X (gfx1250) — matrix operand order repaired and checked by reading, not compiled, not run on hardware
//
#include <hip/hip_runtime.h>
#include <math.h>

// ---- problem constants (from reference setup) ----
#define BB   8
#define CC   64
#define GG   4
#define GC   16          // channels per group
#define HH   128
#define WW   128
#define HWSZ (HH * WW)   // 16384
#define SS   2
#define OC   32          // 2*G*S*S
#define OH   (HH * SS)   // 256
#define OW   (WW * SS)   // 256
#define NT   4           // N-tiles per wave in WMMA GEMM

typedef __attribute__((ext_vector_type(16))) _Float16 v16h;
typedef __attribute__((ext_vector_type(8)))  float    v8f;

__device__ __forceinline__ float sigmoidf_(float v) {
    return 1.0f / (1.0f + __expf(-v));
}

// ------------------------------------------------------------------
// 1) GroupNorm statistics: one block per (b,g), reduce 16*128*128 vals
// ------------------------------------------------------------------
__global__ void gn_stats_kernel(const float* __restrict__ x, float* __restrict__ stats) {
    const int bg = blockIdx.x;                 // 0..31
    const int b = bg / GG, g = bg % GG;
    const float* p = x + ((size_t)b * CC + (size_t)g * GC) * HWSZ;
    const int n = GC * HWSZ;                   // 262144
    float s = 0.f, ss = 0.f;
    for (int i = threadIdx.x; i < n; i += blockDim.x) {
        float v = p[i];
        s += v; ss += v * v;
    }
    __shared__ float shs[256];
    __shared__ float shq[256];
    shs[threadIdx.x] = s; shq[threadIdx.x] = ss;
    __syncthreads();
    for (int off = 128; off > 0; off >>= 1) {
        if ((int)threadIdx.x < off) {
            shs[threadIdx.x] += shs[threadIdx.x + off];
            shq[threadIdx.x] += shq[threadIdx.x + off];
        }
        __syncthreads();
    }
    if (threadIdx.x == 0) {
        float mu  = shs[0] / (float)n;
        float var = shq[0] / (float)n - mu * mu;
        stats[bg * 2 + 0] = mu;
        stats[bg * 2 + 1] = rsqrtf(var + 1e-5f);
    }
}

// ------------------------------------------------------------------
// 2) GroupNorm apply: xn = (x-mu)*rs*gamma + beta  (fp32, NCHW) and
//    also emit K-contiguous f16 transpose xnT[b][pixel][c] for WMMA B.
// ------------------------------------------------------------------
__global__ void gn_apply_kernel(const float* __restrict__ x,
                                const float* __restrict__ gnw,
                                const float* __restrict__ gnb,
                                const float* __restrict__ stats,
                                float* __restrict__ xn,
                                _Float16* __restrict__ xnT) {
    size_t idx = (size_t)blockIdx.x * blockDim.x + threadIdx.x;   // B*C*HW exact
    int pix = (int)(idx % HWSZ);
    int c   = (int)((idx / HWSZ) % CC);
    int b   = (int)(idx / ((size_t)CC * HWSZ));
    int bg  = b * GG + c / GC;
    float mu = stats[bg * 2 + 0];
    float rs = stats[bg * 2 + 1];
    float v = (x[idx] - mu) * rs * gnw[c] + gnb[c];
    xn[idx] = v;
    xnT[((size_t)b * HWSZ + pix) * CC + c] = (_Float16)v;
}

// ------------------------------------------------------------------
// 2b) Pack scope_w (32x64 fp32) into exact WMMA A-fragment lane order:
//     wpk[((mtile*2+kc)*32 + lane)*16 + i]  (one b128-pair load per lane)
// ------------------------------------------------------------------
__global__ void pack_weights_kernel(const float* __restrict__ wgt,
                                    _Float16* __restrict__ wpk) {
    int idx = blockIdx.x * blockDim.x + threadIdx.x;              // 2048 exact
    int i     = idx & 15;
    int lane  = (idx >> 4) & 31;
    int kc    = (idx >> 9) & 1;
    int mtile = (idx >> 10) & 1;
    int mrow  = lane & 15;
    int koffA = (lane >> 4) * 16;               // A: k 0-15 vs 16-31
    int ka = kc * 32 + i + koffA;
    int m  = mtile * 16 + mrow;
    wpk[idx] = (_Float16)wgt[m * 64 + ka];
}

// ------------------------------------------------------------------
// 3) Neighborhood cosine similarity -> soft top-K -> dv (B, 2*G, H, W)
// ------------------------------------------------------------------
__global__ void sim_dv_kernel(const float* __restrict__ xn,
                              const float* __restrict__ raw_tau,
                              float* __restrict__ dv) {
    size_t idx = (size_t)blockIdx.x * blockDim.x + threadIdx.x;   // B*G*HW exact
    int w = (int)(idx % WW);
    int h = (int)((idx / WW) % HH);
    int g = (int)((idx / HWSZ) % GG);
    int b = (int)(idx / ((size_t)GG * HWSZ));

    const float* base = xn + ((size_t)b * CC + (size_t)g * GC) * HWSZ;

    const int di[8] = {-1, -1, -1, 0, 0, 1, 1, 1};
    const int dj[8] = {-1, 0, 1, -1, 1, -1, 0, 1};

    float dot[8], nn2[8];
    float cc = 0.f;
    #pragma unroll
    for (int k = 0; k < 8; ++k) { dot[k] = 0.f; nn2[k] = 0.f; }

    for (int ch = 0; ch < GC; ++ch) {
        const float* pc = base + (size_t)ch * HWSZ;
        float cv = pc[h * WW + w];
        cc += cv * cv;
        #pragma unroll
        for (int k = 0; k < 8; ++k) {
            int hh = h + di[k], ww2 = w + dj[k];
            float nv = (hh >= 0 && hh < HH && ww2 >= 0 && ww2 < WW)
                         ? pc[hh * WW + ww2] : 0.f;
            dot[k] += cv * nv;
            nn2[k] += nv * nv;
        }
    }

    float ncd = fmaxf(sqrtf(cc), 1e-8f);
    float sim[8];
    #pragma unroll
    for (int k = 0; k < 8; ++k)
        sim[k] = dot[k] / (ncd * fmaxf(sqrtf(nn2[k]), 1e-8f));

    // kth = sorted-ascending[8-TOPK] == index 5 (3rd largest)
    float srt[8];
    #pragma unroll
    for (int k = 0; k < 8; ++k) srt[k] = sim[k];
    for (int i = 1; i < 8; ++i) {
        float key = srt[i]; int j = i - 1;
        while (j >= 0 && srt[j] > key) { srt[j + 1] = srt[j]; --j; }
        srt[j + 1] = key;
    }
    float kth = srt[5];
    float tau = sigmoidf_(raw_tau[0]) * 9.0f + 1.0f;

    const float R = 0.70710678118654752f;
    const float d8x[8] = {-R, -1.f, -R, 0.f, 0.f,  R, 1.f, R};
    const float d8y[8] = {-R, 0.f,   R, -1.f, 1.f, -R, 0.f, R};

    float dvx = 0.f, dvy = 0.f;
    #pragma unroll
    for (int k = 0; k < 8; ++k) {
        float m = sigmoidf_((sim[k] - kth) / tau * 20.0f);
        float sw = sim[k] * m;
        dvx += sw * d8x[k];
        dvy += sw * d8y[k];
    }
    size_t hw = (size_t)h * WW + w;
    dv[((size_t)b * (2 * GG) + g) * HWSZ + hw]      = dvx;   // channel d=0 -> g
    dv[((size_t)b * (2 * GG) + GG + g) * HWSZ + hw] = dvy;   // channel d=1 -> G+g
}

// ------------------------------------------------------------------
// 4) 3x3 conv, 8 -> 32 channels, pad=1 (direct): off = conv(dv) + bias
// ------------------------------------------------------------------
__global__ void conv_do_kernel(const float* __restrict__ dv,
                               const float* __restrict__ cw,
                               const float* __restrict__ cb,
                               float* __restrict__ off) {
    size_t idx = (size_t)blockIdx.x * blockDim.x + threadIdx.x;   // B*OC*HW exact
    int w  = (int)(idx % WW);
    int h  = (int)((idx / WW) % HH);
    int oc = (int)((idx / HWSZ) % OC);
    int b  = (int)(idx / ((size_t)OC * HWSZ));

    __builtin_prefetch(&cw[(size_t)oc * 8 * 9], 0, 1);  // global_prefetch_b8

    float acc = cb[oc];
    const float* dvb = dv + (size_t)b * (2 * GG) * HWSZ;
    #pragma unroll
    for (int ic = 0; ic < 8; ++ic) {
        const float* dvc = dvb + (size_t)ic * HWSZ;
        const float* wk  = cw + ((size_t)oc * 8 + ic) * 9;
        #pragma unroll
        for (int ki = 0; ki < 3; ++ki) {
            int hh = h + ki - 1;
            if (hh < 0 || hh >= HH) continue;
            #pragma unroll
            for (int kj = 0; kj < 3; ++kj) {
                int ww2 = w + kj - 1;
                if (ww2 < 0 || ww2 >= WW) continue;
                acc += dvc[hh * WW + ww2] * wk[ki * 3 + kj];
            }
        }
    }
    off[idx] = acc;
}

// ------------------------------------------------------------------
// 5) 1x1 conv 64 -> 32 as WMMA GEMM with packed f16 operands.
//    Each wave: one mtile x 4 ntiles, 8x v_wmma_f32_16x16x32_f16.
//    A: one 32B contiguous fragment/lane; B: one 32B fragment/lane/tile.
// ------------------------------------------------------------------
__global__ void scope_wmma_kernel(const _Float16* __restrict__ xnT, // [B][HW][64]
                                  const _Float16* __restrict__ wpk, // [2][2][32][16]
                                  float* __restrict__ scope) {
    const int ngroups = (BB * HWSZ) / (16 * NT);                  // 2048
    int wave = (int)(((size_t)blockIdx.x * blockDim.x + threadIdx.x) >> 5);
    int lane = threadIdx.x & 31;
    int mtile = wave / ngroups;                                   // 0..1 (uniform)
    int grp   = wave % ngroups;

    int gpix = grp * (16 * NT);          // flat pixel base over B*HW (64-aligned)
    int b    = gpix / HWSZ;
    int pix  = gpix - b * HWSZ;

    int col   = lane & 15;
    int hi    = lane >> 4;               // 0: lanes 0-15, 1: lanes 16-31
    int koffB = hi * 16;                 // B: k 0-15 vs 16-31

    v8f acc[NT];
    #pragma unroll
    for (int t = 0; t < NT; ++t) acc[t] = (v8f){};

    #pragma unroll
    for (int kc = 0; kc < 2; ++kc) {
        v16h av = *(const v16h*)&wpk[(((size_t)mtile * 2 + kc) * 32 + lane) * 16];
        #pragma unroll
        for (int t = 0; t < NT; ++t) {
            size_t n = (size_t)gpix + t * 16 + col;
            v16h bv = *(const v16h*)&xnT[n * CC + kc * 32 + koffB];
            acc[t] = __builtin_amdgcn_wmma_f32_16x16x32_f16(
                         false, av, false, bv, (short)0, acc[t], false, false);
        }
    }

    // C/D layout: lane l, VGPR r -> M = mbase + r, N = l&15
    int mbase = mtile * 16 + 8 * hi;
    float* sp = scope + ((size_t)b * OC + mbase) * HWSZ + pix;
    #pragma unroll
    for (int t = 0; t < NT; ++t) {
        #pragma unroll
        for (int r = 0; r < 8; ++r)
            sp[(size_t)r * HWSZ + t * 16 + col] = acc[t][r];
    }
}

// ------------------------------------------------------------------
// 6) offset assembly + pixel-shuffle coords + bilinear grid sample
// ------------------------------------------------------------------
__global__ void sample_kernel(const float* __restrict__ xn,
                              const float* __restrict__ off,
                              const float* __restrict__ scope,
                              float* __restrict__ out) {
    size_t idx = (size_t)blockIdx.x * blockDim.x + threadIdx.x;   // B*G*OH*OW exact
    int ow = (int)(idx % OW);
    int oh = (int)((idx / OW) % OH);
    int g  = (int)((idx / ((size_t)OH * OW)) % GG);
    int b  = (int)(idx / ((size_t)GG * OH * OW));

    int h = oh >> 1, si = oh & 1;
    int w = ow >> 1, sj = ow & 1;
    size_t hw = (size_t)h * WW + w;

    float coords[2];
    #pragma unroll
    for (int d = 0; d < 2; ++d) {
        int chan = d * 16 + g * 4 + si * 2 + sj;
        size_t p = ((size_t)b * OC + chan) * HWSZ + hw;
        float ov = off[p];
        float sv = scope[p];
        float ip = ((d == 0 ? sj : si) ? 0.25f : -0.25f);
        float o  = ov * sigmoidf_(sv) * 0.5f + ip;
        float bs = (d == 0) ? ((float)w + 0.5f) : ((float)h + 0.5f);
        float nm = (d == 0) ? (float)WW : (float)HH;
        coords[d] = 2.0f * (bs + o) / nm - 1.0f;
    }

    float xf = (coords[0] + 1.0f) * 0.5f * (float)WW - 0.5f;
    float yf = (coords[1] + 1.0f) * 0.5f * (float)HH - 0.5f;
    float x0 = floorf(xf), y0 = floorf(yf);
    float wx = xf - x0, wy = yf - y0;
    int x0i = min(max((int)x0, 0), WW - 1);
    int x1i = min(max((int)x0 + 1, 0), WW - 1);
    int y0i = min(max((int)y0, 0), HH - 1);
    int y1i = min(max((int)y0 + 1, 0), HH - 1);

    const float* img = xn + ((size_t)b * CC + (size_t)g * GC) * HWSZ;
    float* op = out + (((size_t)b * CC + (size_t)g * GC) * OH + oh) * OW + ow;

    #pragma unroll
    for (int ch = 0; ch < GC; ++ch) {
        const float* ic = img + (size_t)ch * HWSZ;
        float v00 = ic[y0i * WW + x0i];
        float v01 = ic[y0i * WW + x1i];
        float v10 = ic[y1i * WW + x0i];
        float v11 = ic[y1i * WW + x1i];
        float v = (v00 * (1.f - wx) + v01 * wx) * (1.f - wy)
                + (v10 * (1.f - wx) + v11 * wx) * wy;
        op[(size_t)ch * OH * OW] = v;
    }
}

// ------------------------------------------------------------------
// host-side launcher
// ------------------------------------------------------------------
extern "C" void kernel_launch(void* const* d_in, const int* in_sizes, int n_in,
                              void* d_out, int out_size, void* d_ws, size_t ws_size,
                              hipStream_t stream) {
    (void)in_sizes; (void)n_in; (void)out_size; (void)ws_size;

    const float* x       = (const float*)d_in[0];
    const float* gnw     = (const float*)d_in[1];
    const float* gnb     = (const float*)d_in[2];
    const float* raw_tau = (const float*)d_in[3];
    const float* cdw     = (const float*)d_in[4];
    const float* cdb     = (const float*)d_in[5];
    const float* sw      = (const float*)d_in[6];
    float* out = (float*)d_out;

    // workspace layout (floats; every region 64-float aligned)
    float* ws      = (float*)d_ws;
    float*    stats = ws;                                        // 64
    float*    xn    = stats + 64;                                // 8388608
    float*    dv    = xn + (size_t)BB * CC * HWSZ;               // 1048576
    float*    off   = dv + (size_t)BB * 2 * GG * HWSZ;           // 4194304
    float*    scope = off + (size_t)BB * OC * HWSZ;              // 4194304
    _Float16* wpk   = (_Float16*)(scope + (size_t)BB * OC * HWSZ); // 2048 halfs
    _Float16* xnT   = wpk + 2048;                                // 8388608 halfs

    // 1) GN stats: 32 blocks
    gn_stats_kernel<<<BB * GG, 256, 0, stream>>>(x, stats);

    // 2) GN apply + f16 transpose: 32768 blocks
    gn_apply_kernel<<<(BB * CC * HWSZ) / 256, 256, 0, stream>>>(x, gnw, gnb, stats, xn, xnT);

    // 2b) weight packing: 2048 elements = 8 blocks
    pack_weights_kernel<<<8, 256, 0, stream>>>(sw, wpk);

    // 3) similarity -> dv: 2048 blocks
    sim_dv_kernel<<<(BB * GG * HWSZ) / 256, 256, 0, stream>>>(xn, raw_tau, dv);

    // 4) 3x3 conv: 16384 blocks
    conv_do_kernel<<<(BB * OC * HWSZ) / 256, 256, 0, stream>>>(dv, cdw, cdb, off);

    // 5) 1x1 conv via WMMA: 2 mtiles * 2048 groups = 4096 waves = 512 blocks
    scope_wmma_kernel<<<512, 256, 0, stream>>>(xnT, wpk, scope);

    // 6) sample: 8192 blocks
    sample_kernel<<<(BB * GG * OH * OW) / 256, 256, 0, stream>>>(xn, off, scope, out);
}
